// MambaModel_84439057039722
// MI455X (gfx1250) — matrix operand order repaired and checked by reading, not compiled, not run on hardware
//
#include <hip/hip_runtime.h>

// ---------------- model constants ----------------
static constexpr int Bc      = 8;
static constexpr int Lc      = 2048;
static constexpr int D_IN    = 64;
static constexpr int D_MODEL = 128;
static constexpr int D_OUT   = 16;
static constexpr int N_LAYERS= 2;
static constexpr int D_INNER = 256;
static constexpr int D_STATE = 16;
static constexpr int D_CONV  = 4;
static constexpr int DT_RANK = 8;
static constexpr int Mrows   = Bc * Lc;        // 16384 token rows

typedef __attribute__((ext_vector_type(16))) _Float16 v16h;
typedef __attribute__((ext_vector_type(8)))  float    v8f;

__device__ __forceinline__ float silu_f(float x) {
    return x / (1.0f + __expf(-x));
}
__device__ __forceinline__ float softplus_f(float x) {
    return (x > 20.0f) ? x : log1pf(__expf(x));
}
__device__ __forceinline__ float4 ld4(const float* p) {
    return *(const float4*)p;
}

// ---------------------------------------------------------------------------
// Fully-templated WMMA GEMM:  C[M,N] = act( A[M,K] * W[N,K]^T + bias )
//
// fp32 in memory, f16 operands on the fly, f32 accumulate via
// v_wmma_f32_16x16x32_f16.  One wave per 16x16 output tile; the 8 waves of a
// block share one N-tile, so for regular shapes the 16xK weight tile is
// staged once per block into LDS with gfx1250 async global->LDS DMA
// (global_load_async_to_lds_b128, ASYNCcnt) and B fragments are then read
// from LDS (ds_load_b128).  Tail shapes (N=40, K=8) keep direct global loads
// so nothing reads out of bounds.  ACT: 0 = none, 1 = softplus.
// ---------------------------------------------------------------------------
template<int Kt, int Nt, int LDA, int LDC, int ACT, bool HASBIAS>
__global__ __launch_bounds__(256)
void wmma_gemm(const float* __restrict__ A,
               const float* __restrict__ W,    // [Nt,Kt] row-major
               const float* __restrict__ bias, // used iff HASBIAS
               float* __restrict__ C)
{
    constexpr int  Mtiles = Mrows >> 4;
    constexpr bool STAGE  = (Kt % 32 == 0) && ((Nt & 15) == 0);

    const int wave = blockIdx.x * (blockDim.x >> 5) + (threadIdx.x >> 5);
    const int tm   = wave % Mtiles;
    const int tn   = wave / Mtiles;     // uniform across the block (8 | Mtiles)
    const int lane = threadIdx.x & 31;
    const int hs   = lane >> 4;         // half-wave select
    const int lr   = lane & 15;
    const int row  = (tm << 4) + lr;    // A row owned by this lane
    const int col  = (tn << 4) + lr;    // B/C column owned by this lane

    const float* __restrict__ arow = A + (size_t)row * LDA;
    v8f acc = {};

    if constexpr (STAGE) {
        // ---- stage the 16xKt weight tile into LDS via async DMA ----------
        __shared__ float smW[16 * Kt];
        const unsigned ldsBase =
            (unsigned)(size_t)(__attribute__((address_space(3))) float*)smW;
        const float* gsrc = W + (size_t)tn * 16 * Kt;   // contiguous tile
        constexpr int CHUNKS = (16 * Kt) / 4;           // 16B chunks
        #pragma unroll
        for (int c0 = 0; c0 < CHUNKS; c0 += 256) {
            int c = c0 + threadIdx.x;
            unsigned ldsOff = ldsBase + (unsigned)c * 16u;
            const float* g  = gsrc + (size_t)c * 4;
            asm volatile("global_load_async_to_lds_b128 %0, %1, off"
                         :: "v"(ldsOff), "v"(g) : "memory");
        }
        asm volatile("s_wait_asynccnt 0x0" ::: "memory");
        __syncthreads();

        const float* __restrict__ wl = smW + (size_t)lr * Kt;  // lane's column
        #pragma unroll
        for (int k0 = 0; k0 < Kt; k0 += 32) {
            // A 16x32 f16 fragment: lanes 0-15 K 0-15,
            // lanes 16-31 K 16-31.
            float4 a0 = ld4(arow + k0 + hs * 16);
            float4 a1 = ld4(arow + k0 + hs * 16 + 4);
            float4 a2 = ld4(arow + k0 + hs * 16 + 8);
            float4 a3 = ld4(arow + k0 + hs * 16 + 12);
            // B 32x16 f16 fragment from LDS: lane = column,
            // half-wave = K 0-15|16-31.
            float4 b0 = ld4(wl + k0 + hs * 16);
            float4 b1 = ld4(wl + k0 + hs * 16 + 4);
            float4 b2 = ld4(wl + k0 + hs * 16 + 8);
            float4 b3 = ld4(wl + k0 + hs * 16 + 12);

            const float av[16] = {a0.x,a0.y,a0.z,a0.w, a1.x,a1.y,a1.z,a1.w,
                                  a2.x,a2.y,a2.z,a2.w, a3.x,a3.y,a3.z,a3.w};
            const float bv[16] = {b0.x,b0.y,b0.z,b0.w, b1.x,b1.y,b1.z,b1.w,
                                  b2.x,b2.y,b2.z,b2.w, b3.x,b3.y,b3.z,b3.w};
            v16h afrag, bfrag;
            #pragma unroll
            for (int e = 0; e < 16; ++e) {
                afrag[e] = (_Float16)av[e];
                bfrag[e] = (_Float16)bv[e];
            }
            acc = __builtin_amdgcn_wmma_f32_16x16x32_f16(
                      false, afrag, false, bfrag, (short)0, acc, false, false);
        }
    } else if constexpr (Kt % 32 == 0) {
        // ---- N-tail path (N=40): clamp weight row so tail lanes read
        // in-bounds garbage that is never stored. ------------------------
        const int colc = (col < Nt) ? col : (Nt - 1);
        const float* __restrict__ wrow = W + (size_t)colc * Kt;
        #pragma unroll
        for (int k0 = 0; k0 < Kt; k0 += 32) {
            float4 a0 = ld4(arow + k0 + hs * 16);
            float4 a1 = ld4(arow + k0 + hs * 16 + 4);
            float4 a2 = ld4(arow + k0 + hs * 16 + 8);
            float4 a3 = ld4(arow + k0 + hs * 16 + 12);
            float4 b0 = ld4(wrow + k0 + hs * 16);
            float4 b1 = ld4(wrow + k0 + hs * 16 + 4);
            float4 b2 = ld4(wrow + k0 + hs * 16 + 8);
            float4 b3 = ld4(wrow + k0 + hs * 16 + 12);

            const float av[16] = {a0.x,a0.y,a0.z,a0.w, a1.x,a1.y,a1.z,a1.w,
                                  a2.x,a2.y,a2.z,a2.w, a3.x,a3.y,a3.z,a3.w};
            const float bv[16] = {b0.x,b0.y,b0.z,b0.w, b1.x,b1.y,b1.z,b1.w,
                                  b2.x,b2.y,b2.z,b2.w, b3.x,b3.y,b3.z,b3.w};
            v16h afrag, bfrag;
            #pragma unroll
            for (int e = 0; e < 16; ++e) {
                afrag[e] = (_Float16)av[e];
                bfrag[e] = (_Float16)bv[e];
            }
            acc = __builtin_amdgcn_wmma_f32_16x16x32_f16(
                      false, afrag, false, bfrag, (short)0, acc, false, false);
        }
    } else {
        // ---- K == 8 (dt_proj): valid elements are half-wave 0, e=0..7.
        // A rows sit inside lda=LDA (>=16) x_dbl rows, so both float4 loads
        // are in-bounds; W rows are exactly 8 floats. --------------------
        static_assert(Kt == 8, "tail path is specialized for K == 8");
        const float* __restrict__ wrow = W + (size_t)col * Kt;
        float4 a0 = ld4(arow);       // k 0..3
        float4 a1 = ld4(arow + 4);   // k 4..7
        float4 b0 = ld4(wrow);
        float4 b1 = ld4(wrow + 4);
        if (hs) {                    // half-wave 1 would cover k 8..15 -> zero
            a0.x=a0.y=a0.z=a0.w=0.f; a1.x=a1.y=a1.z=a1.w=0.f;
            b0.x=b0.y=b0.z=b0.w=0.f; b1.x=b1.y=b1.z=b1.w=0.f;
        }
        const float av[8] = {a0.x,a0.y,a0.z,a0.w, a1.x,a1.y,a1.z,a1.w};
        const float bv[8] = {b0.x,b0.y,b0.z,b0.w, b1.x,b1.y,b1.z,b1.w};
        v16h afrag = {}, bfrag = {};
        #pragma unroll
        for (int e = 0; e < 8; ++e) {
            afrag[e] = (_Float16)av[e];
            bfrag[e] = (_Float16)bv[e];
        }
        acc = __builtin_amdgcn_wmma_f32_16x16x32_f16(
                  false, afrag, false, bfrag, (short)0, acc, false, false);
    }

    // C/D layout: vgpr i -> M = i + 8*hs, N = lr.
    float bv = 0.0f;
    if constexpr (HASBIAS) bv = bias[(Nt & 15) ? ((col < Nt) ? col : 0) : col];
    if ((Nt & 15) == 0 || col < Nt) {
        #pragma unroll
        for (int i = 0; i < 8; ++i) {
            int r = (tm << 4) + hs * 8 + i;
            float v = acc[i] + bv;
            if constexpr (ACT == 1) v = softplus_f(v);
            C[(size_t)r * LDC + col] = v;
        }
    }
}

template<int Kt, int Nt, int LDA, int LDC, int ACT, bool HASBIAS>
static inline void launch_gemm(const float* A, const float* W, const float* bias,
                               float* C, hipStream_t stream)
{
    constexpr int Ntiles = (Nt + 15) >> 4;
    constexpr int blocks = (Mrows >> 4) * Ntiles / 8;   // exact: 8 waves/block
    wmma_gemm<Kt, Nt, LDA, LDC, ACT, HASBIAS>
        <<<blocks, 256, 0, stream>>>(A, W, bias, C);
}

// ---------------------------------------------------------------------------
// Depthwise causal conv1d (width 4) + bias + SiLU.
// ---------------------------------------------------------------------------
__global__ void conv_silu_kernel(const float* __restrict__ xz,  // [M,512]
                                 const float* __restrict__ cw,  // [256,4]
                                 const float* __restrict__ cb,  // [256]
                                 float* __restrict__ xc)        // [M,256]
{
    int idx = blockIdx.x * blockDim.x + threadIdx.x;   // over M*256
    if (idx >= Mrows * D_INNER) return;
    int d  = idx & (D_INNER - 1);
    int bl = idx >> 8;
    int l  = bl & (Lc - 1);
    int b  = bl >> 11;
    float s = cb[d];
    #pragma unroll
    for (int j = 0; j < D_CONV; ++j) {
        int lj = l - (D_CONV - 1) + j;
        if (lj >= 0)
            s += cw[d * D_CONV + j] * xz[((size_t)(b * Lc + lj) << 9) + d];
    }
    xc[idx] = silu_f(s);
}

// ---------------------------------------------------------------------------
// Selective scan, fused with C-contraction, D skip-connection and SiLU gate.
// One thread per (b, d, n); 16 states of one channel on 16 consecutive lanes
// of a wave32 -> state reduction via 4 shfl_xor steps.
// ---------------------------------------------------------------------------
__global__ void ssm_scan_kernel(const float* __restrict__ dt,    // [M,256]
                                const float* __restrict__ xdbl,  // [M,40]
                                const float* __restrict__ xc,    // [M,256]
                                const float* __restrict__ xz,    // [M,512] (res)
                                const float* __restrict__ A_log, // [256,16]
                                const float* __restrict__ Dp,    // [256]
                                float* __restrict__ y)           // [M,256]
{
    int gid = blockIdx.x * blockDim.x + threadIdx.x;   // B*256*16 = 32768
    int n = gid & (D_STATE - 1);
    int d = (gid >> 4) & (D_INNER - 1);
    int b = gid >> 12;

    const float Areg = -__expf(A_log[d * D_STATE + n]);
    const float Dreg = Dp[d];
    float hst = 0.0f;
    const size_t base = (size_t)b * Lc;

    for (int l = 0; l < Lc; ++l) {
        size_t t = base + l;
        if (l + 8 < Lc) {   // uniform across wave -> no EXEC split
            __builtin_prefetch(&dt[(t + 8) * D_INNER + d], 0, 0);
            __builtin_prefetch(&xdbl[(t + 8) * 40 + n], 0, 0);
        }
        float dtv = dt[t * D_INNER + d];
        float xcv = xc[t * D_INNER + d];
        float Bv  = xdbl[t * 40 + DT_RANK + n];
        float Cv  = xdbl[t * 40 + DT_RANK + D_STATE + n];

        float dA = __expf(dtv * Areg);
        hst = dA * hst + dtv * Bv * xcv;

        float p = hst * Cv;
        p += __shfl_xor(p, 1, 32);
        p += __shfl_xor(p, 2, 32);
        p += __shfl_xor(p, 4, 32);
        p += __shfl_xor(p, 8, 32);

        if (n == 0) {
            float res = xz[(t << 9) + D_INNER + d];
            y[t * D_INNER + d] = (p + xcv * Dreg) * silu_f(res);
        }
    }
}

// ---------------------------------------------------------------------------
__global__ void pool_kernel(const float* __restrict__ h, float* __restrict__ pooled)
{
    int b = blockIdx.x, dm = threadIdx.x;
    float s = 0.0f;
    for (int l = 0; l < Lc; ++l)
        s += h[((size_t)(b * Lc + l)) * D_MODEL + dm];
    pooled[b * D_MODEL + dm] = s * (1.0f / Lc);
}

__global__ void fc_kernel(const float* __restrict__ pooled,
                          const float* __restrict__ fcw,
                          const float* __restrict__ fcb,
                          float* __restrict__ out)
{
    int t = threadIdx.x;
    if (t >= Bc * D_OUT) return;
    int b = t >> 4, o = t & (D_OUT - 1);
    float s = fcb[o];
    #pragma unroll 8
    for (int k = 0; k < D_MODEL; ++k)
        s += pooled[b * D_MODEL + k] * fcw[o * D_MODEL + k];
    out[b * D_OUT + o] = s;
}

// ---------------------------------------------------------------------------
extern "C" void kernel_launch(void* const* d_in, const int* in_sizes, int n_in,
                              void* d_out, int out_size, void* d_ws, size_t ws_size,
                              hipStream_t stream)
{
    const float* x       = (const float*)d_in[0];   // [B,L,64]
    const float* proj_w  = (const float*)d_in[1];   // [128,64]
    const float* proj_b  = (const float*)d_in[2];   // [128]
    const float* in_w    = (const float*)d_in[3];   // [2,512,128]
    const float* conv_w  = (const float*)d_in[4];   // [2,256,1,4]
    const float* conv_b  = (const float*)d_in[5];   // [2,256]
    const float* xw      = (const float*)d_in[6];   // [2,40,256]
    const float* dtw     = (const float*)d_in[7];   // [2,256,8]
    const float* dtbias  = (const float*)d_in[8];   // [2,256]
    const float* A_log   = (const float*)d_in[9];   // [2,256,16]
    const float* Dp      = (const float*)d_in[10];  // [2,256]
    const float* out_w   = (const float*)d_in[11];  // [2,128,256]
    const float* fc_w    = (const float*)d_in[12];  // [16,128]
    const float* fc_b    = (const float*)d_in[13];  // [16]
    float* out = (float*)d_out;

    // workspace layout (fp32 elements)
    float* ws     = (float*)d_ws;
    float* h      = ws;                                   // M*128
    float* xz     = h    + (size_t)Mrows * D_MODEL;       // M*512
    float* xc     = xz   + (size_t)Mrows * 2 * D_INNER;   // M*256
    float* xdbl   = xc   + (size_t)Mrows * D_INNER;       // M*40
    float* dtbuf  = xdbl + (size_t)Mrows * 40;            // M*256
    float* y      = dtbuf+ (size_t)Mrows * D_INNER;       // M*256
    float* pooled = y    + (size_t)Mrows * D_INNER;       // B*128

    // 1) input projection: h = x @ proj_w^T + proj_b
    launch_gemm<64, 128, 64, 128, 0, true>(x, proj_w, proj_b, h, stream);

    for (int layer = 0; layer < N_LAYERS; ++layer) {
        const float* lw_in  = in_w   + (size_t)layer * 2 * D_INNER * D_MODEL;
        const float* lw_cw  = conv_w + (size_t)layer * D_INNER * D_CONV;
        const float* lw_cb  = conv_b + (size_t)layer * D_INNER;
        const float* lw_xw  = xw     + (size_t)layer * (DT_RANK + 2 * D_STATE) * D_INNER;
        const float* lw_dtw = dtw    + (size_t)layer * D_INNER * DT_RANK;
        const float* lw_dtb = dtbias + (size_t)layer * D_INNER;
        const float* lw_Al  = A_log  + (size_t)layer * D_INNER * D_STATE;
        const float* lw_Dp  = Dp     + (size_t)layer * D_INNER;
        const float* lw_ow  = out_w  + (size_t)layer * D_MODEL * D_INNER;

        // 2) in_proj: xz = h @ in_w^T                      [M,512]
        launch_gemm<128, 512, 128, 512, 0, false>(h, lw_in, nullptr, xz, stream);

        // 3) depthwise causal conv + SiLU                  [M,256]
        {
            int n = Mrows * D_INNER;
            conv_silu_kernel<<<(n + 255) / 256, 256, 0, stream>>>(xz, lw_cw, lw_cb, xc);
        }

        // 4) x_proj: xdbl = xc @ xw^T                      [M,40] (dt|B|C)
        launch_gemm<256, 40, 256, 40, 0, false>(xc, lw_xw, nullptr, xdbl, stream);

        // 5) dt_proj + softplus (K=8 tail path)            [M,256]
        launch_gemm<8, 256, 40, 256, 1, true>(xdbl, lw_dtw, lw_dtb, dtbuf, stream);

        // 6) fused selective scan + gate                   [M,256]
        ssm_scan_kernel<<<(Bc * D_INNER * D_STATE) / 256, 256, 0, stream>>>(
            dtbuf, xdbl, xc, xz, lw_Al, lw_Dp, y);

        // 7) out_proj: h = y @ out_w^T                     [M,128]
        launch_gemm<256, 128, 256, 128, 0, false>(y, lw_ow, nullptr, h, stream);
    }

    // 8) mean-pool + FC head
    pool_kernel<<<Bc, D_MODEL, 0, stream>>>(h, pooled);
    fc_kernel<<<1, 128, 0, stream>>>(pooled, fc_w, fc_b, out);
}
